// GAT_86148454023888
// MI455X (gfx1250) — compile-verified, matrix operand order checked by reading, not run on hardware
//
#include <hip/hip_runtime.h>
#include <hip/hip_bf16.h>
#include <cstdint>

// ---------------- constants (match reference) ----------------
#define N_NODES 50000
#define N_EDGES 600000
#define F_IN 128
#define C 128
#define LIN 256
#define OUT_DIM 10
#define N_GRAPHS 128
#define NEG_SLOPE 0.2f
#define E_TOT (N_EDGES + N_NODES)

typedef __attribute__((ext_vector_type(16))) __bf16 bf16x16;
typedef __attribute__((ext_vector_type(8)))  __bf16 bf16x8;
typedef __attribute__((ext_vector_type(8)))  float  f32x8;

// ---- ordered-uint encoding for float atomic max (monotonic) ----
__device__ __forceinline__ unsigned fenc(float f) {
    unsigned u = __float_as_uint(f);
    return (u & 0x80000000u) ? ~u : (u | 0x80000000u);
}
__device__ __forceinline__ float fdec(unsigned e) {
    unsigned v = (e & 0x80000000u) ? (e & 0x7FFFFFFFu) : ~e;
    return __uint_as_float(v);
}
__device__ __forceinline__ unsigned short f2bf(float f) {
    unsigned u = __float_as_uint(f);
    return (unsigned short)((u + 0x7FFFu + ((u >> 16) & 1u)) >> 16);  // RNE
}

// ---------------- elementwise helpers ----------------
__global__ void k_f32_to_bf16(const float* __restrict__ src,
                              unsigned short* __restrict__ dst, int n) {
    int i = blockIdx.x * blockDim.x + threadIdx.x;
    if (i < n) dst[i] = f2bf(src[i]);
}

__global__ void k_fill_u32(unsigned* __restrict__ p, unsigned v, int n) {
    int i = blockIdx.x * blockDim.x + threadIdx.x;
    if (i < n) p[i] = v;
}

// ---------------- pack f32 B[K,N] into bf16 WMMA B-fragment layout -------------
// Fragment order per (nt, kb) tile: lane L holds 16 contiguous bf16:
//   j<8 : k = kb*32 +      (L/16)*8 + j       n = nt*16 + (L%16)
//   j>=8: k = kb*32 + 16 + (L/16)*8 + (j-8)
// Packed index = ((nt*(K/32) + kb)*32 + L)*16 + j  -> GEMM loads two b128 / lane.
__global__ void k_pack_B_bf16(const float* __restrict__ B,
                              unsigned short* __restrict__ Bp, int K, int N) {
    int i = blockIdx.x * blockDim.x + threadIdx.x;
    if (i >= K * N) return;
    int j  = i & 15;
    int L  = (i >> 4) & 31;
    int tl = i >> 9;                 // tile linear = nt*(K/32) + kb
    int kbcnt = K >> 5;
    int kb = tl % kbcnt;
    int nt = tl / kbcnt;
    int k  = kb * 32 + ((j < 8) ? ((L >> 4) * 8 + j) : (16 + (L >> 4) * 8 + (j - 8)));
    int n  = nt * 16 + (L & 15);
    Bp[i] = f2bf(B[(size_t)k * N + n]);
}

// ---------------- WMMA bf16 GEMM: C[M,N] = A[M,K] @ Bp (+bias, relu) -----------
// One wave computes one 16x(16*NT) strip: A fragment reused across NT WMMAs per
// 32-wide k-step. A is natural row-major bf16 (two b128 loads/lane), B comes
// from the pre-packed fragment buffer (two b128 loads/lane/tile).
template <int NT, bool BIAS_RELU>
__global__ void k_gemm_wmma_bf16(const unsigned short* __restrict__ Ab,
                                 const unsigned short* __restrict__ Bp,
                                 float* __restrict__ Co,
                                 int M, int N, int K,
                                 const float* __restrict__ bias) {
    const int lane = threadIdx.x & 31;
    const int wave = threadIdx.x >> 5;
    const int ngrpN = (N >> 4) / NT;
    const int grp = blockIdx.x * (blockDim.x >> 5) + wave;
    const int ntot = (M >> 4) * ngrpN;
    if (grp >= ntot) return;              // wave-uniform: EXEC stays all-1s
    const int mt = grp / ngrpN, ng = grp % ngrpN;
    const int hf  = lane >> 4;
    const int l16 = lane & 15;
    const int arow = mt * 16 + l16;
    const int kbcnt = K >> 5;

    const __bf16* A = (const __bf16*)Ab;
    const __bf16* B = (const __bf16*)Bp;

    f32x8 acc[NT];
    #pragma unroll
    for (int t = 0; t < NT; ++t) acc[t] = (f32x8){};

    for (int kb = 0; kb < kbcnt; ++kb) {
        const int k0 = kb * 32;
        __builtin_prefetch(A + (size_t)arow * K + k0 + 32, 0, 1);  // global_prefetch_b8
        bf16x8 a0 = *(const bf16x8*)(A + (size_t)arow * K + k0 + hf * 8);
        bf16x8 a1 = *(const bf16x8*)(A + (size_t)arow * K + k0 + 16 + hf * 8);
        bf16x16 av;
        #pragma unroll
        for (int j = 0; j < 8; ++j) { av[j] = a0[j]; av[8 + j] = a1[j]; }
        #pragma unroll
        for (int t = 0; t < NT; ++t) {
            const int nt = ng * NT + t;
            const __bf16* bp = B + (((size_t)nt * kbcnt + kb) * 32 + lane) * 16;
            bf16x8 b0 = *(const bf16x8*)(bp);
            bf16x8 b1 = *(const bf16x8*)(bp + 8);
            bf16x16 bv;
            #pragma unroll
            for (int j = 0; j < 8; ++j) { bv[j] = b0[j]; bv[8 + j] = b1[j]; }
            acc[t] = __builtin_amdgcn_wmma_f32_16x16x32_bf16(
                         false, av, false, bv, (short)0, acc[t], false, false);
        }
    }
    // C/D layout: VGPR r -> lanes 0-15: M=r, lanes 16-31: M=8+r; N = lane%16
    #pragma unroll
    for (int t = 0; t < NT; ++t) {
        const int nbase = (ng * NT + t) * 16 + l16;
        #pragma unroll
        for (int r = 0; r < 8; ++r) {
            const int m = mt * 16 + r + hf * 8;
            float v = acc[t][r];
            if (BIAS_RELU) {
                v += bias[nbase];
                v = v > 0.f ? v : 0.f;
            }
            Co[(size_t)m * N + nbase] = v;
        }
    }
}

// ---------------- attention logits: als[i]=h[i].asrc, ald[i]=h[i].adst ----------
__global__ void k_compute_al(const float* __restrict__ h,
                             const float* __restrict__ asrc,
                             const float* __restrict__ adst,
                             float* __restrict__ als, float* __restrict__ ald,
                             int n) {
    int wid  = (blockIdx.x * blockDim.x + threadIdx.x) >> 5;
    int lane = threadIdx.x & 31;
    if (wid >= n) return;
    float4 v = ((const float4*)(h + (size_t)wid * C))[lane];
    float4 s = ((const float4*)asrc)[lane];
    float4 d = ((const float4*)adst)[lane];
    float ps = v.x * s.x + v.y * s.y + v.z * s.z + v.w * s.w;
    float pd = v.x * d.x + v.y * d.y + v.z * d.z + v.w * d.w;
    #pragma unroll
    for (int off = 16; off; off >>= 1) {
        ps += __shfl_down(ps, off);
        pd += __shfl_down(pd, off);
    }
    if (lane == 0) { als[wid] = ps; ald[wid] = pd; }
}

// ---------------- edge pass 1: segment max of leaky-relu scores ----------------
__global__ void k_edge_max(const int* __restrict__ ei,
                           const float* __restrict__ als,
                           const float* __restrict__ ald,
                           unsigned* __restrict__ menc) {
    int e = blockIdx.x * blockDim.x + threadIdx.x;
    if (e >= E_TOT) return;
    int s, d;
    if (e < N_EDGES) { s = ei[e]; d = ei[N_EDGES + e]; } else { s = d = e - N_EDGES; }
    float v = als[s] + ald[d];
    v = v > 0.f ? v : NEG_SLOPE * v;
    atomicMax(menc + d, fenc(v));
}

__global__ void k_decode_m(const unsigned* __restrict__ menc,
                           float* __restrict__ mval, int n) {
    int i = blockIdx.x * blockDim.x + threadIdx.x;
    if (i >= n) return;
    unsigned e = menc[i];
    float m = (e == 0u) ? 0.f : fdec(e);
    mval[i] = isfinite(m) ? m : 0.f;
}

// ---------------- edge pass 2: exp + denominator -------------------------------
__global__ void k_edge_exp(const int* __restrict__ ei,
                           const float* __restrict__ als,
                           const float* __restrict__ ald,
                           const float* __restrict__ mval,
                           float* __restrict__ exbuf,
                           float* __restrict__ denom) {
    int e = blockIdx.x * blockDim.x + threadIdx.x;
    if (e >= E_TOT) return;
    int s, d;
    if (e < N_EDGES) { s = ei[e]; d = ei[N_EDGES + e]; } else { s = d = e - N_EDGES; }
    float v = als[s] + ald[d];
    v = v > 0.f ? v : NEG_SLOPE * v;
    float ex = __expf(v - mval[d]);
    exbuf[e] = ex;
    atomicAdd(denom + d, ex);
}

// ---------------- edge pass 3: scatter alpha * h[src] into agg[dst] ------------
__global__ void k_edge_scatter(const int* __restrict__ ei,
                               const float* __restrict__ exbuf,
                               const float* __restrict__ denom,
                               const float* __restrict__ h,
                               float* __restrict__ agg) {
    int gid  = blockIdx.x * blockDim.x + threadIdx.x;
    int wid  = gid >> 5;
    int lane = threadIdx.x & 31;
    if (wid >= E_TOT) return;
    int s, d;
    if (wid < N_EDGES) { s = ei[wid]; d = ei[N_EDGES + wid]; } else { s = d = wid - N_EDGES; }
    float dn = denom[d];
    dn = dn > 1e-16f ? dn : 1e-16f;
    float alpha = exbuf[wid] / dn;
    float4 hv = ((const float4*)(h + (size_t)s * C))[lane];
    float* o = agg + (size_t)d * C + lane * 4;
    atomicAdd(o + 0, alpha * hv.x);
    atomicAdd(o + 1, alpha * hv.y);
    atomicAdd(o + 2, alpha * hv.z);
    atomicAdd(o + 3, alpha * hv.w);
}

// ---------------- bias + relu --------------------------------------------------
__global__ void k_bias_relu(const float* __restrict__ agg,
                            const float* __restrict__ b,
                            float* __restrict__ xo, int n) {
    int i = blockIdx.x * blockDim.x + threadIdx.x;
    if (i >= n) return;
    float v = agg[i] + b[i & (C - 1)];
    xo[i] = v > 0.f ? v : 0.f;
}

// ---------------- global max pool ----------------------------------------------
__global__ void k_pool_max(const float* __restrict__ x,
                           const int* __restrict__ batch,
                           unsigned* __restrict__ genc) {
    int gid = blockIdx.x * blockDim.x + threadIdx.x;
    if (gid >= N_NODES * C) return;
    int node = gid >> 7, f = gid & (C - 1);
    atomicMax(genc + batch[node] * C + f, fenc(x[gid]));
}

__global__ void k_decode_g(const unsigned* __restrict__ genc,
                           float* __restrict__ g, int n) {
    int i = blockIdx.x * blockDim.x + threadIdx.x;
    if (i >= n) return;
    unsigned e = genc[i];
    float v = (e == 0u) ? 0.f : fdec(e);
    g[i] = isfinite(v) ? v : 0.f;
}

// ---------------- output layer: [128,256] @ [256,10] + b -----------------------
__global__ void k_out_layer(const float* __restrict__ lin_out,
                            const float* __restrict__ W,
                            const float* __restrict__ b,
                            float* __restrict__ out) {
    int gid = blockIdx.x * blockDim.x + threadIdx.x;
    if (gid >= N_GRAPHS * OUT_DIM) return;
    int g = gid / OUT_DIM, o = gid % OUT_DIM;
    float acc = b[o];
    for (int k = 0; k < LIN; ++k) acc += lin_out[g * LIN + k] * W[k * OUT_DIM + o];
    out[gid] = acc;
}

// ================================================================================
extern "C" void kernel_launch(void* const* d_in, const int* in_sizes, int n_in,
                              void* d_out, int out_size, void* d_ws, size_t ws_size,
                              hipStream_t stream) {
    const float* x0     = (const float*)d_in[0];
    const int*   ei     = (const int*)d_in[2];
    const int*   batch  = (const int*)d_in[3];
    const float* Wl[3]  = {(const float*)d_in[4], (const float*)d_in[8],  (const float*)d_in[12]};
    const float* asr[3] = {(const float*)d_in[5], (const float*)d_in[9],  (const float*)d_in[13]};
    const float* ads[3] = {(const float*)d_in[6], (const float*)d_in[10], (const float*)d_in[14]};
    const float* bl[3]  = {(const float*)d_in[7], (const float*)d_in[11], (const float*)d_in[15]};
    const float* lin_W  = (const float*)d_in[16];
    const float* lin_b  = (const float*)d_in[17];
    const float* out_W  = (const float*)d_in[18];
    const float* out_b  = (const float*)d_in[19];

    // ---- workspace carve (256B aligned) ----
    char* w = (char*)d_ws;
    auto carve = [&](size_t bytes) -> void* {
        void* p = (void*)w;
        w += (bytes + 255) & ~(size_t)255;
        return p;
    };
    unsigned short* xb    = (unsigned short*)carve((size_t)N_NODES * C * 2);
    unsigned short* wb    = (unsigned short*)carve((size_t)C * LIN * 2);  // packed weights
    float*          h     = (float*)carve((size_t)N_NODES * C * 4);
    float*          agg   = (float*)carve((size_t)N_NODES * C * 4);
    float*          xcur  = (float*)carve((size_t)N_NODES * C * 4);
    float*          als   = (float*)carve((size_t)N_NODES * 4);
    float*          ald   = (float*)carve((size_t)N_NODES * 4);
    unsigned*       menc  = (unsigned*)carve((size_t)N_NODES * 4);
    float*          mval  = (float*)carve((size_t)N_NODES * 4);
    float*          denom = (float*)carve((size_t)N_NODES * 4);
    float*          exbuf = (float*)carve((size_t)E_TOT * 4);
    unsigned*       genc  = (unsigned*)carve((size_t)N_GRAPHS * C * 4);
    float*          gf    = (float*)carve((size_t)N_GRAPHS * C * 4);
    unsigned short* gb    = (unsigned short*)carve((size_t)N_GRAPHS * C * 2);
    float*          lout  = (float*)carve((size_t)N_GRAPHS * LIN * 4);

    const int BLK = 256;
    auto nb = [](long long n, int blk) { return (int)((n + blk - 1) / blk); };

    const float* xin = x0;
    for (int l = 0; l < 3; ++l) {
        const int NC = N_NODES * C;
        // bf16 conversions: x natural row-major, W packed into fragment layout
        k_f32_to_bf16<<<nb(NC, BLK), BLK, 0, stream>>>(xin, xb, NC);
        k_pack_B_bf16<<<nb(C * C, BLK), BLK, 0, stream>>>(Wl[l], wb, C, C);
        // h = x @ W  (WMMA bf16, f32 accum); 1 wave -> 16x64 strip (NT=4)
        {
            int grps = (N_NODES / 16) * ((C / 16) / 4);   // 3125 * 2 = 6250
            k_gemm_wmma_bf16<4, false><<<nb(grps, 8), BLK, 0, stream>>>(
                xb, wb, h, N_NODES, C, C, nullptr);
        }
        // attention logits
        k_compute_al<<<nb((long long)N_NODES * 32, BLK), BLK, 0, stream>>>(
            h, asr[l], ads[l], als, ald, N_NODES);
        // init accumulators
        k_fill_u32<<<nb(N_NODES, BLK), BLK, 0, stream>>>(menc, 0u, N_NODES);
        k_fill_u32<<<nb(N_NODES, BLK), BLK, 0, stream>>>((unsigned*)denom, 0u, N_NODES);
        k_fill_u32<<<nb(NC, BLK), BLK, 0, stream>>>((unsigned*)agg, 0u, NC);
        // softmax-attention over edges
        k_edge_max<<<nb(E_TOT, BLK), BLK, 0, stream>>>(ei, als, ald, menc);
        k_decode_m<<<nb(N_NODES, BLK), BLK, 0, stream>>>(menc, mval, N_NODES);
        k_edge_exp<<<nb(E_TOT, BLK), BLK, 0, stream>>>(ei, als, ald, mval, exbuf, denom);
        k_edge_scatter<<<nb((long long)E_TOT * 32, BLK), BLK, 0, stream>>>(
            ei, exbuf, denom, h, agg);
        // bias + relu
        k_bias_relu<<<nb(NC, BLK), BLK, 0, stream>>>(agg, bl[l], xcur, NC);
        xin = xcur;
    }

    // global max pool
    k_fill_u32<<<nb(N_GRAPHS * C, BLK), BLK, 0, stream>>>(genc, 0u, N_GRAPHS * C);
    k_pool_max<<<nb((long long)N_NODES * C, BLK), BLK, 0, stream>>>(xcur, batch, genc);
    k_decode_g<<<nb(N_GRAPHS * C, BLK), BLK, 0, stream>>>(genc, gf, N_GRAPHS * C);

    // lin layer (WMMA): [128,128] @ [128,256] + b, relu
    k_f32_to_bf16<<<nb(N_GRAPHS * C, BLK), BLK, 0, stream>>>(gf, gb, N_GRAPHS * C);
    k_pack_B_bf16<<<nb(C * LIN, BLK), BLK, 0, stream>>>(lin_W, wb, C, LIN);
    {
        int grps = (N_GRAPHS / 16) * ((LIN / 16) / 4);    // 8 * 4 = 32
        k_gemm_wmma_bf16<4, true><<<nb(grps, 8), BLK, 0, stream>>>(
            gb, wb, lout, N_GRAPHS, LIN, C, lin_b);
    }
    // output layer
    k_out_layer<<<nb(N_GRAPHS * OUT_DIM, BLK), BLK, 0, stream>>>(
        lout, out_W, out_b, (float*)d_out);
}
